// DPQ_56977036149389
// MI455X (gfx1250) — matrix operand order repaired and checked by reading, not compiled, not run on hardware
//
#include <hip/hip_runtime.h>

// ---------------------------------------------------------------------------
// DPQ forward for MI455X (gfx1250, wave32, WMMA).
// Pipeline: cvt/transpose prep -> WMMA GEMM (encoder, 16x64 strip/wave) ->
// fused dot/softmax/lookup (WMMA + wave32 shuffle softmax + gumbel prefetch)
// -> WMMA GEMM (decoder). norm1 cancels inside softmax, never computed.
// ---------------------------------------------------------------------------

#define NROWS 32768
#define DDIM  512
#define MSUB  8
#define KCODE 512
#define DSUBV 64
// TAU == 1.0f (folded into nt / gumbel add)

typedef __attribute__((ext_vector_type(16))) __bf16 v16bf;
typedef __attribute__((ext_vector_type(8)))  float  v8f;

union ABfrag {
    v16bf        v;
    unsigned int u[8];
};

__device__ __forceinline__ unsigned short f2bf(float f) {
    union { float f; unsigned int u; } x; x.f = f;
    unsigned int u = x.u;
    // round-to-nearest-even bf16
    u = (u + 0x7FFFu + ((u >> 16) & 1u)) >> 16;
    return (unsigned short)u;
}

// ---------------- prep kernels ---------------------------------------------

__global__ void k_cvt_bf16(const float* __restrict__ in,
                           unsigned short* __restrict__ out, int n) {
    int i = blockIdx.x * blockDim.x + threadIdx.x;
    if (i < n) out[i] = f2bf(in[i]);
}

// W [512][512] row-major (row = reduction dim) -> WT[col][k] bf16
__global__ void k_build_wT(const float* __restrict__ W,
                           unsigned short* __restrict__ WT) {
    int i = blockIdx.x * blockDim.x + threadIdx.x;
    if (i < DDIM * DDIM) {
        int k = i >> 9, c = i & 511;
        WT[c * DDIM + k] = f2bf(W[i]);
    }
}

// codebook [M][K][64] -> cb_bf (same layout) and cbT [M][64][K]
__global__ void k_build_cb(const float* __restrict__ cb,
                           unsigned short* __restrict__ cb_bf,
                           unsigned short* __restrict__ cbT) {
    int i = blockIdx.x * blockDim.x + threadIdx.x;
    if (i < MSUB * KCODE * DSUBV) {
        int d = i & 63, k = (i >> 6) & 511, m = i >> 15;
        unsigned short v = f2bf(cb[i]);
        cb_bf[i] = v;
        cbT[((m * DSUBV + d) << 9) + k] = v;
    }
}

// nt[m][k] = -||c_mk||^2 * exp(-logT[m]) / TAU
__global__ void k_nt(const float* __restrict__ cb,
                     const float* __restrict__ logt,
                     float* __restrict__ nt) {
    int i = blockIdx.x * blockDim.x + threadIdx.x;
    if (i < MSUB * KCODE) {
        int m = i >> 9;
        const float* r = cb + (size_t)i * DSUBV;
        float s = 0.f;
        #pragma unroll
        for (int d = 0; d < DSUBV; ++d) s += r[d] * r[d];
        nt[i] = -s * __expf(-logt[m]);
    }
}

// ---------------- WMMA GEMM: C[N,512] = A[N,512] @ B[512,512] + bias --------
// A row-major bf16, BT = B^T row-major bf16 (B-frag K-pairs contiguous).
// One wave computes a 16x64 output strip: 4 C tiles share one A-fragment
// per k-step (4x less A traffic, 4 WMMAs back-to-back per k-step).

template <bool OUT_BF16>
__global__ void k_gemm512(const unsigned short* __restrict__ A,
                          const unsigned short* __restrict__ BT,
                          const float* __restrict__ bias,
                          void* __restrict__ out) {
    const int row0 = blockIdx.x * 16;
    const int col0 = blockIdx.y * 64;     // 4 tiles of 16 columns
    const int L    = threadIdx.x;         // 0..31 (wave32)
    const int col  = L & 15;
    const int half = L >> 4;

    v8f acc[4];
    #pragma unroll
    for (int t = 0; t < 4; ++t) {
        float b = bias[col0 + t * 16 + col];  // C layout: N = L&15 in all VGPRs
        #pragma unroll
        for (int v = 0; v < 8; ++v) acc[t][v] = b;
    }

    const unsigned short* arow = A + (size_t)(row0 + (L & 15)) * DDIM;
    const unsigned short* bcol = BT + (size_t)(col0 + col) * DDIM;

    #pragma unroll 2
    for (int kt = 0; kt < DDIM / 32; ++kt) {
        ABfrag af;
        #pragma unroll
        for (int v = 0; v < 8; ++v) {
            int ka = kt * 32 + ((v & 3) * 2) + ((v >= 4) ? 16 : 0) + half * 8;
            af.u[v] = *(const unsigned int*)(arow + ka);
        }
        #pragma unroll
        for (int t = 0; t < 4; ++t) {
            ABfrag bf;
            #pragma unroll
            for (int v = 0; v < 8; ++v) {
                int kb = kt * 32 + ((v & 3) * 2) + ((v >= 4) ? 16 : 0) + half * 8;
                bf.u[v] = *(const unsigned int*)(bcol + (size_t)t * 16 * DDIM + kb);
            }
            acc[t] = __builtin_amdgcn_wmma_f32_16x16x32_bf16(
                         false, af.v, false, bf.v, (short)0, acc[t], false, false);
        }
    }

    #pragma unroll
    for (int t = 0; t < 4; ++t) {
        #pragma unroll
        for (int v = 0; v < 8; ++v) {
            int r = row0 + v + (half ? 8 : 0);
            size_t idx = (size_t)r * DDIM + col0 + t * 16 + col;
            if (OUT_BF16) ((unsigned short*)out)[idx] = f2bf(acc[t][v]);
            else          ((float*)out)[idx]          = acc[t][v];
        }
    }
}

// ---------------- fused quantize: dot -> softmax -> codebook lookup --------
// one wave per (16-row block, subspace m)

__global__ void k_quantize(const unsigned short* __restrict__ h_bf,  // [N][512]
                           const unsigned short* __restrict__ cb_bf, // [M][K][64]
                           const unsigned short* __restrict__ cbT,   // [M][64][K]
                           const float* __restrict__ nt,             // [M][K]
                           const float* __restrict__ logt,           // [M]
                           const float* __restrict__ gumbel,         // [N][M][K]
                           unsigned short* __restrict__ recon_bf) {  // [N][512]
    __shared__ float          sc[16][KCODE];      // 32 KB scores
    __shared__ unsigned short codes[16][KCODE];   // 16 KB bf16 codes

    const int n0   = blockIdx.x * 16;
    const int m    = blockIdx.y;
    const int L    = threadIdx.x;
    const int col  = L & 15;
    const int half = L >> 4;

    // ---- A-fragments of h for this 16x64 subspace slice (k = 0..63) -------
    const unsigned short* hrow = h_bf + (size_t)(n0 + (L & 15)) * DDIM + m * DSUBV;
    ABfrag a0, a1;
    #pragma unroll
    for (int v = 0; v < 8; ++v) {
        int k = ((v & 3) * 2) + ((v >= 4) ? 16 : 0) + half * 8;
        a0.u[v] = *(const unsigned int*)(hrow + k);
        a1.u[v] = *(const unsigned int*)(hrow + 32 + k);
    }

    // warm the gumbel stream for the first two rows (global_prefetch_b8)
    {
        const float* g0 = gumbel + ((size_t)n0 * MSUB + m) * KCODE;
        const float* g1 = gumbel + ((size_t)(n0 + 1) * MSUB + m) * KCODE;
        __builtin_prefetch(g0 + L * 16, 0, 1);    // 32 lanes x 64B = 2KB row
        __builtin_prefetch(g1 + L * 16, 0, 1);
    }

    // ---- dot[n, k] = h . c_k   (32 tiles of 16 codes, K-reduction = 64) ---
    for (int kt = 0; kt < KCODE / 16; ++kt) {
        const unsigned short* crow = cb_bf + ((size_t)(m * KCODE + kt * 16 + col)) * DSUBV;
        ABfrag b0, b1;
        #pragma unroll
        for (int v = 0; v < 8; ++v) {
            int d = ((v & 3) * 2) + ((v >= 4) ? 16 : 0) + half * 8;
            b0.u[v] = *(const unsigned int*)(crow + d);
            b1.u[v] = *(const unsigned int*)(crow + 32 + d);
        }
        v8f acc = {};
        acc = __builtin_amdgcn_wmma_f32_16x16x32_bf16(
                  false, a0.v, false, b0.v, (short)0, acc, false, false);
        acc = __builtin_amdgcn_wmma_f32_16x16x32_bf16(
                  false, a1.v, false, b1.v, (short)0, acc, false, false);
        #pragma unroll
        for (int v = 0; v < 8; ++v)
            sc[v + (half ? 8 : 0)][kt * 16 + col] = acc[v];
    }
    __syncthreads();

    // ---- Gumbel-softmax per row over K=512 (norm1 cancels) ----------------
    const float amul = 2.0f * __expf(-logt[m]);   // 2*invT/TAU
    const float* ntm = nt + (size_t)m * KCODE;

    for (int r = 0; r < 16; ++r) {
        const float* grow = gumbel + ((size_t)(n0 + r) * MSUB + m) * KCODE;
        if (r + 2 < 16) {                         // stay 2 rows ahead
            const float* gnext = gumbel + ((size_t)(n0 + r + 2) * MSUB + m) * KCODE;
            __builtin_prefetch(gnext + L * 16, 0, 1);
        }
        float lg[16];
        float mx = -3.4e38f;
        #pragma unroll
        for (int i = 0; i < 16; ++i) {
            int c = L + 32 * i;                   // coalesced gumbel stream
            float val = sc[r][c] * amul + ntm[c] + grow[c];
            lg[i] = val;
            mx = fmaxf(mx, val);
        }
        #pragma unroll
        for (int off = 16; off >= 1; off >>= 1)
            mx = fmaxf(mx, __shfl_xor(mx, off, 32));
        float sum = 0.f;
        #pragma unroll
        for (int i = 0; i < 16; ++i) {
            float e = __expf(lg[i] - mx);
            lg[i] = e;
            sum += e;
        }
        #pragma unroll
        for (int off = 16; off >= 1; off >>= 1)
            sum += __shfl_xor(sum, off, 32);
        float inv = 1.0f / sum;
        #pragma unroll
        for (int i = 0; i < 16; ++i)
            codes[r][L + 32 * i] = f2bf(lg[i] * inv);
    }
    __syncthreads();

    // ---- recon[16,64] = codes[16,512] @ codebook[m][512,64] ---------------
    for (int t = 0; t < 4; ++t) {                 // 4 column tiles of 16
        v8f acc = {};
        for (int kt = 0; kt < KCODE / 32; ++kt) {
            ABfrag af, bf;
            #pragma unroll
            for (int v = 0; v < 8; ++v) {
                int ka = kt * 32 + ((v & 3) * 2) + ((v >= 4) ? 16 : 0) + half * 8;
                af.u[v] = *(const unsigned int*)&codes[L & 15][ka];
                int kb = kt * 32 + ((v & 3) * 2) + ((v >= 4) ? 16 : 0) + half * 8;
                bf.u[v] = *(const unsigned int*)(cbT + ((size_t)(m * DSUBV + t * 16 + col)) * KCODE + kb);
            }
            acc = __builtin_amdgcn_wmma_f32_16x16x32_bf16(
                      false, af.v, false, bf.v, (short)0, acc, false, false);
        }
        #pragma unroll
        for (int v = 0; v < 8; ++v) {
            int r = n0 + v + (half ? 8 : 0);
            recon_bf[(size_t)r * DDIM + m * DSUBV + t * 16 + col] = f2bf(acc[v]);
        }
    }
}

// ---------------------------------------------------------------------------

extern "C" void kernel_launch(void* const* d_in, const int* in_sizes, int n_in,
                              void* d_out, int out_size, void* d_ws, size_t ws_size,
                              hipStream_t stream) {
    const float* x      = (const float*)d_in[0];
    const float* W_enc  = (const float*)d_in[1];
    const float* b_enc  = (const float*)d_in[2];
    const float* cb     = (const float*)d_in[3];
    const float* logt   = (const float*)d_in[4];
    const float* gumbel = (const float*)d_in[5];
    const float* W_dec  = (const float*)d_in[6];
    const float* b_dec  = (const float*)d_in[7];
    float* out = (float*)d_out;

    // workspace carve-out (~103 MB)
    char* ws = (char*)d_ws;
    size_t off = 0;
    auto carve = [&](size_t bytes) {
        char* p = ws + off;
        off = (off + bytes + 255) & ~(size_t)255;
        return p;
    };
    unsigned short* x_bf     = (unsigned short*)carve((size_t)NROWS * DDIM * 2);
    unsigned short* h_bf     = (unsigned short*)carve((size_t)NROWS * DDIM * 2);
    unsigned short* recon_bf = (unsigned short*)carve((size_t)NROWS * DDIM * 2);
    unsigned short* WencT    = (unsigned short*)carve((size_t)DDIM * DDIM * 2);
    unsigned short* WdecT    = (unsigned short*)carve((size_t)DDIM * DDIM * 2);
    unsigned short* cb_bf    = (unsigned short*)carve((size_t)MSUB * KCODE * DSUBV * 2);
    unsigned short* cbT      = (unsigned short*)carve((size_t)MSUB * KCODE * DSUBV * 2);
    float*          nt       = (float*)carve((size_t)MSUB * KCODE * 4);
    (void)off; (void)ws_size; (void)in_sizes; (void)n_in; (void)out_size;

    // prep
    {
        int n = NROWS * DDIM;
        k_cvt_bf16<<<(n + 255) / 256, 256, 0, stream>>>(x, x_bf, n);
        k_build_wT<<<(DDIM * DDIM + 255) / 256, 256, 0, stream>>>(W_enc, WencT);
        k_build_wT<<<(DDIM * DDIM + 255) / 256, 256, 0, stream>>>(W_dec, WdecT);
        k_build_cb<<<(MSUB * KCODE * DSUBV + 255) / 256, 256, 0, stream>>>(cb, cb_bf, cbT);
        k_nt<<<(MSUB * KCODE + 255) / 256, 256, 0, stream>>>(cb, logt, nt);
    }

    // encoder GEMM: h = x @ W_enc + b_enc  (bf16 out)
    {
        dim3 grid(NROWS / 16, DDIM / 64);
        k_gemm512<true><<<grid, 32, 0, stream>>>(x_bf, WencT, b_enc, (void*)h_bf);
    }

    // fused distance / gumbel-softmax / codebook lookup
    {
        dim3 grid(NROWS / 16, MSUB);
        k_quantize<<<grid, 32, 0, stream>>>(h_bf, cb_bf, cbT, nt, logt, gumbel, recon_bf);
    }

    // decoder GEMM: out = recon @ W_dec + b_dec  (f32 out)
    {
        dim3 grid(NROWS / 16, DDIM / 64);
        k_gemm512<false><<<grid, 32, 0, stream>>>(recon_bf, WdecT, b_dec, (void*)out);
    }
}
